// AppearanceDecoder_63419487092826
// MI455X (gfx1250) — matrix operand order repaired and checked by reading, not compiled, not run on hardware
//
#include <hip/hip_runtime.h>
#include <hip/hip_bf16.h>
#include <cstdint>

// ---------------------------------------------------------------------------
// Types / WMMA helpers (CDNA5 gfx1250, wave32)
// ---------------------------------------------------------------------------
typedef __attribute__((ext_vector_type(16))) __bf16 v16bf;
typedef __attribute__((ext_vector_type(8)))  float  v8f;
typedef __attribute__((ext_vector_type(4)))  __bf16 v4bf;

union FragU { v16bf v; unsigned int u[8]; };

#define QP   112      // Q=100 padded to 7 x 16
#define MT   7        // m-tiles per block
#define HID  256
#define BNUM 8
#define CTOT 1792     // 256+512+1024

__device__ __forceinline__ v8f wmma_bf16(v16bf a, v16bf b, v8f c) {
  return __builtin_amdgcn_wmma_f32_16x16x32_bf16(false, a, false, b, (short)0, c,
                                                 false, false);
}

// Async global->LDS 16-byte copy (CDNA5, ASYNCcnt-tracked, no VGPR round trip).
// dst = LDS byte address (low 32 bits of flat shared address), src = global.
__device__ __forceinline__ void async_b128(unsigned lds_addr, const void* gaddr) {
  asm volatile("global_load_async_to_lds_b128 %0, %1, off"
               :: "v"(lds_addr), "v"(gaddr) : "memory");
}
__device__ __forceinline__ void wait_async0() {
  asm volatile("s_wait_asynccnt 0x0" ::: "memory");
}
__device__ __forceinline__ unsigned lds_addr_of(const void* p) {
  return (unsigned)(uintptr_t)p;   // LDS aperture: addr[31:0] = LDS offset
}

// A fragment 16x32 bf16 from row-major [m][k] (k contiguous). ISA layout:
// lane m = lane&15, half h = lane>>4; VGPR r holds K = 2r + 8h + (r>=4 ? 8 : 0).
__device__ __forceinline__ v16bf fragA(const __bf16* base, int lda, int lane) {
  const __bf16* p = base + (size_t)(lane & 15) * lda;
  const int h8 = (lane >> 4) * 8;
  FragU f;
#pragma unroll
  for (int r = 0; r < 8; ++r) {
    int k = 2 * r + h8 + ((r >= 4) ? 8 : 0);
    f.u[r] = *(const unsigned int*)(p + k);
  }
  return f.v;
}

// A fragment built from an fp32 row pointer (row already selected per lane).
__device__ __forceinline__ v16bf fragA_f32(const float* p, int lane) {
  const int h8 = (lane >> 4) * 8;
  v16bf a;
#pragma unroll
  for (int r = 0; r < 8; ++r) {
    int k = 2 * r + h8 + ((r >= 4) ? 8 : 0);
    float2 t = *(const float2*)(p + k);
    a[2 * r]     = (__bf16)t.x;
    a[2 * r + 1] = (__bf16)t.y;
  }
  return a;
}

// B fragment 32x16 bf16 where the data for column n is k-contiguous.
// p = column base for this lane (caller applied n = lane&15).
// lane half h: VGPR r holds K = 2r + 8h + (r>=4 ? 8 : 0).
__device__ __forceinline__ v16bf fragB_kcontig(const __bf16* p, int lane) {
  const int h16 = (lane >> 4) * 8;
  FragU f;
#pragma unroll
  for (int r = 0; r < 8; ++r) f.u[r] = *(const unsigned int*)(p + 2 * r + h16 + ((r >= 4) ? 8 : 0));
  return f.v;
}

// B fragment from row-major [k][n] tile (n contiguous): base at (k=0, n0).
__device__ __forceinline__ v16bf fragB_ncontig(const __bf16* base, int ld, int lane) {
  const __bf16* p = base + (lane & 15);
  const int h16 = (lane >> 4) * 16;
  v16bf b;
#pragma unroll
  for (int r = 0; r < 8; ++r) {
    b[2 * r]     = p[(size_t)(2 * r + h16 / 2 + ((r >= 4) ? 8 : 0)) * ld];
    b[2 * r + 1] = p[(size_t)(2 * r + 1 + h16 / 2 + ((r >= 4) ? 8 : 0)) * ld];
  }
  return b;
}

// ---------------------------------------------------------------------------
// Prep kernels
// ---------------------------------------------------------------------------
__global__ void cvt_f32_to_bf16(const float* __restrict__ in,
                                __bf16* __restrict__ out, int n4) {
  int i = blockIdx.x * 256 + threadIdx.x;
  if (i >= n4) return;
  float4 v = ((const float4*)in)[i];
  v4bf o = {(__bf16)v.x, (__bf16)v.y, (__bf16)v.z, (__bf16)v.w};
  *(v4bf*)(out + 4 * i) = o;
}

__global__ void zero_f32(float* p, int n) {
  int i = blockIdx.x * 256 + threadIdx.x;
  if (i < n) p[i] = 0.0f;
}

// output [Q=100][B=8][256] fp32 -> padded [B][112][256] bf16 (zeros for q>=100)
__global__ void pad_output_bf16(const float* __restrict__ outp,
                                __bf16* __restrict__ obf) {
  int i = blockIdx.x * 256 + threadIdx.x;
  if (i >= BNUM * QP * HID) return;
  int c  = i & 255;
  int bq = i >> 8;
  int q  = bq % QP;
  int b  = bq / QP;
  float v = (q < 100) ? outp[((size_t)q * BNUM + b) * HID + c] : 0.0f;
  obf[i] = (__bf16)v;
}

// concat conv weights into [256][1792] bf16
__global__ void build_wcat(const float* __restrict__ w0, const float* __restrict__ w1,
                           const float* __restrict__ w2, __bf16* __restrict__ wcat) {
  int i = blockIdx.x * 256 + threadIdx.x;
  if (i >= HID * CTOT) return;
  int c = i / CTOT, ig = i - c * CTOT;
  float v;
  if (ig < 256)       v = w0[c * 256  + ig];
  else if (ig < 768)  v = w1[c * 512  + (ig - 256)];
  else                v = w2[c * 1024 + (ig - 768)];
  wcat[i] = (__bf16)v;
}

// ---------------------------------------------------------------------------
// Generic WMMA GEMM: C[112 x 128-tile] per block, 4 waves x 2 n-subtiles.
// Tiles staged to LDS with double-buffered GLOBAL_LOAD_ASYNC_TO_LDS_B128.
//   BT=0: B(k,n) = Bm[k*ldb + n]   (n contiguous)  -> LDS [32k][128n]
//   BT=1: B(k,n) = Bm[n*ldb + k]   (k contiguous)  -> LDS [128n][32k]
//   OUTMODE=0: bf16 store; OUTMODE=1: f32 atomicAdd (split-K via grid.z)
// ---------------------------------------------------------------------------
template <int BT, int OUTMODE>
__global__ __launch_bounds__(128) void gemm_qp(
    const __bf16* __restrict__ Ab, long long As, int lda,
    const __bf16* __restrict__ Bb, long long Bs, int ldb,
    int Kseg, void* __restrict__ Ob, long long Os, int ldo) {
  __shared__ __align__(16) __bf16 At[2][QP * 32];
  __shared__ __align__(16) __bf16 Bt[2][4096];

  const int tid  = threadIdx.x;
  const int lane = tid & 31;
  const int wave = tid >> 5;
  const int batch = blockIdx.y;
  const int n0 = blockIdx.x * 128;
  const int kz = blockIdx.z * Kseg;

  const __bf16* A  = Ab + (size_t)batch * As;
  const __bf16* Bm = Bb + (size_t)batch * Bs;

  const v8f vzero = {0.f, 0.f, 0.f, 0.f, 0.f, 0.f, 0.f, 0.f};
  v8f acc[MT][2];
#pragma unroll
  for (int mt = 0; mt < MT; ++mt) { acc[mt][0] = vzero; acc[mt][1] = vzero; }

  auto issue_tiles = [&](int chunk, int buf) {
    const int kg = kz + chunk * 32;
    // A tile: 112 x 32 bf16 = 448 x B128
    for (int e = tid; e < 448; e += 128) {
      int row = e >> 2, cc = (e & 3) * 8;
      async_b128(lds_addr_of(&At[buf][row * 32 + cc]),
                 &A[(size_t)row * lda + kg + cc]);
    }
    if (BT) {  // 128 n-rows x 32 k = 512 x B128
      for (int e = tid; e < 512; e += 128) {
        int row = e >> 2, cc = (e & 3) * 8;
        async_b128(lds_addr_of(&Bt[buf][row * 32 + cc]),
                   &Bm[(size_t)(n0 + row) * ldb + kg + cc]);
      }
    } else {   // 32 k-rows x 128 n = 512 x B128
      for (int e = tid; e < 512; e += 128) {
        int row = e >> 4, cc = (e & 15) * 8;
        async_b128(lds_addr_of(&Bt[buf][row * 128 + cc]),
                   &Bm[(size_t)(kg + row) * ldb + n0 + cc]);
      }
    }
  };

  const int nchunks = Kseg / 32;
  issue_tiles(0, 0);
  for (int ic = 0; ic < nchunks; ++ic) {
    const int buf = ic & 1;
    wait_async0();        // my async writes to LDS are done
    __syncthreads();      // everyone's are done
    if (ic + 1 < nchunks) issue_tiles(ic + 1, buf ^ 1);  // prefetch next chunk

    const __bf16* at = At[buf];
    const __bf16* bt = Bt[buf];
    v16bf bf0, bf1;
    if (BT) {
      bf0 = fragB_kcontig(&bt[(wave * 32 + (lane & 15)) * 32], lane);
      bf1 = fragB_kcontig(&bt[(wave * 32 + 16 + (lane & 15)) * 32], lane);
    } else {
      bf0 = fragB_ncontig(&bt[wave * 32], 128, lane);
      bf1 = fragB_ncontig(&bt[wave * 32 + 16], 128, lane);
    }
#pragma unroll
    for (int mt = 0; mt < MT; ++mt) {
      v16bf af = fragA(&at[mt * 16 * 32], 32, lane);
      acc[mt][0] = wmma_bf16(af, bf0, acc[mt][0]);
      acc[mt][1] = wmma_bf16(af, bf1, acc[mt][1]);
    }
    __syncthreads();      // done reading buf before it is refilled
  }

  // C layout: lane n = lane&15, rows m = r + 8*(lane>>4)
  const int n  = lane & 15;
  const int mb = (lane >> 4) * 8;
#pragma unroll
  for (int nt = 0; nt < 2; ++nt) {
    const int col = n0 + wave * 32 + nt * 16 + n;
    if (OUTMODE == 0) {
      __bf16* O = (__bf16*)Ob + (size_t)batch * Os;
#pragma unroll
      for (int mt = 0; mt < MT; ++mt)
#pragma unroll
        for (int r = 0; r < 8; ++r)
          O[(size_t)(mt * 16 + mb + r) * ldo + col] = (__bf16)acc[mt][nt][r];
    } else {
      float* O = (float*)Ob + (size_t)batch * Os;
#pragma unroll
      for (int mt = 0; mt < MT; ++mt)
#pragma unroll
        for (int r = 0; r < 8; ++r)
          atomicAdd(&O[(size_t)(mt * 16 + mb + r) * ldo + col], acc[mt][nt][r]);
    }
  }
}

// ---------------------------------------------------------------------------
// In-place row softmax over bf16 scores [B*QP rows][D]
// ---------------------------------------------------------------------------
__global__ __launch_bounds__(256) void softmax_rows(__bf16* __restrict__ S, int D) {
  __shared__ float red[256];
  const int q = blockIdx.x, b = blockIdx.y, tid = threadIdx.x;
  __bf16* row = S + (size_t)(b * QP + q) * D;

  float mx = -1e30f;
  for (int d = tid; d < D; d += 256) mx = fmaxf(mx, (float)row[d]);
  red[tid] = mx; __syncthreads();
  for (int s = 128; s > 0; s >>= 1) {
    if (tid < s) red[tid] = fmaxf(red[tid], red[tid + s]);
    __syncthreads();
  }
  mx = red[0]; __syncthreads();

  float sm = 0.f;
  for (int d = tid; d < D; d += 256) sm += __expf((float)row[d] - mx);
  red[tid] = sm; __syncthreads();
  for (int s = 128; s > 0; s >>= 1) {
    if (tid < s) red[tid] += red[tid + s];
    __syncthreads();
  }
  const float inv = 1.0f / red[0];
  __syncthreads();
  for (int d = tid; d < D; d += 256)
    row[d] = (__bf16)(__expf((float)row[d] - mx) * inv);
}

// ---------------------------------------------------------------------------
// Tail: back-project av -> concat[16x768] -> agg MLP -> LN -> proj MLP -> out
// One block of 128 threads (4 waves) handles 16 output rows (r = q*8 + b).
// ---------------------------------------------------------------------------
__device__ __forceinline__ void layer16(int tid, const __bf16* xb, int xstride,
                                        int Kdim, const __bf16* W, int wld,
                                        const float* bias, int relu, float* yf) {
  const int lane = tid & 31, wave = tid >> 5;
  const v8f vzero = {0.f, 0.f, 0.f, 0.f, 0.f, 0.f, 0.f, 0.f};
  for (int t = wave; t < 16; t += 4) {
    const int c0 = t << 4;
    v8f acc = vzero;
    const __bf16* wrow = W + (size_t)(c0 + (lane & 15)) * wld;
    for (int kc = 0; kc < Kdim; kc += 32) {
      v16bf af = fragA(xb + kc, xstride, lane);
      v16bf bf_ = fragB_kcontig(wrow + kc, lane);
      acc = wmma_bf16(af, bf_, acc);
    }
    const int n = lane & 15, mb = (lane >> 4) * 8;
    const float bs = bias[c0 + n];
#pragma unroll
    for (int r = 0; r < 8; ++r) {
      float v = acc[r] + bs;
      if (relu) v = fmaxf(v, 0.0f);
      yf[(mb + r) * 256 + c0 + n] = v;
    }
  }
}

__device__ __forceinline__ void cvt256(int tid, const float* yf, __bf16* xbf) {
  for (int e = tid; e < 16 * 256; e += 128) {
    int m = e >> 8, c = e & 255;
    xbf[m * 776 + c] = (__bf16)yf[e];
  }
}

__global__ __launch_bounds__(128) void tail_kernel(
    const float* __restrict__ av, const __bf16* __restrict__ wcat,
    const float* b0, const float* b1, const float* b2,
    const __bf16* agg1w, const float* agg1b,
    const __bf16* agg2w, const float* agg2b,
    const float* lng, const float* lnb,
    const __bf16* p1w, const float* pb1,
    const __bf16* p2w, const float* pb2,
    const __bf16* p3w, const float* pb3,
    float* __restrict__ out) {
  __shared__ __align__(16) float  xf[16 * 768];
  __shared__ __align__(16) __bf16 xbf[16 * 776];
  __shared__ float red2[256];
  __shared__ float mv[32];

  const int tid = threadIdx.x, lane = tid & 31, wave = tid >> 5;
  const int r0 = blockIdx.x * 16;

  // per-lane av row: r = q*8 + b ; av stored [b][QP][1792]
  const int rr = r0 + (lane & 15);
  const int qa = rr >> 3, ba = rr & 7;
  const float* avrow = av + (size_t)(ba * QP + qa) * CTOT;

  const int CINS[3] = {256, 512, 1024};
  const int COFF[3] = {0, 256, 768};
  const v8f vzero = {0.f, 0.f, 0.f, 0.f, 0.f, 0.f, 0.f, 0.f};

  // Step 1: x[m, l*256+c] = av_l[row] @ w_l^T + conv bias  (48 tiles)
  for (int t = wave; t < 48; t += 4) {
    const int l = t >> 4, c0 = (t & 15) << 4;
    const float* bl = (l == 0) ? b0 : ((l == 1) ? b1 : b2);
    v8f acc = vzero;
    const __bf16* wrow = wcat + (size_t)(c0 + (lane & 15)) * CTOT + COFF[l];
    const float* arow = avrow + COFF[l];
    for (int kc = 0; kc < CINS[l]; kc += 32) {
      v16bf af = fragA_f32(arow + kc, lane);
      v16bf bf_ = fragB_kcontig(wrow + kc, lane);
      acc = wmma_bf16(af, bf_, acc);
    }
    const int n = lane & 15, mb = (lane >> 4) * 8;
    const float bs = bl[c0 + n];
#pragma unroll
    for (int r = 0; r < 8; ++r)
      xf[(mb + r) * 768 + (l << 8) + c0 + n] = acc[r] + bs;
  }
  __syncthreads();
  for (int m = 0; m < 16; ++m)
    for (int c = tid; c < 768; c += 128) xbf[m * 776 + c] = (__bf16)xf[m * 768 + c];
  __syncthreads();

  // agg layer 1 (768 -> 256, relu)
  layer16(tid, xbf, 776, 768, agg1w, 768, agg1b, 1, xf);
  __syncthreads();
  cvt256(tid, xf, xbf);
  __syncthreads();

  // agg layer 2 (256 -> 256)
  layer16(tid, xbf, 776, 256, agg2w, 256, agg2b, 0, xf);
  __syncthreads();

  // LayerNorm over 256 per row (fp32), apply gamma/beta -> bf16
  {
    const int row = tid >> 3, sub = tid & 7;
    float s = 0.f, ss = 0.f;
    for (int c = sub * 32; c < sub * 32 + 32; ++c) {
      float v = xf[row * 256 + c];
      s += v; ss += v * v;
    }
    red2[tid] = s; red2[128 + tid] = ss;
    __syncthreads();
    if (sub == 0) {
      float S = 0.f, SS = 0.f;
      for (int k = 0; k < 8; ++k) { S += red2[row * 8 + k]; SS += red2[128 + row * 8 + k]; }
      float mu = S * (1.0f / 256.0f);
      float var = SS * (1.0f / 256.0f) - mu * mu;
      mv[row * 2] = mu; mv[row * 2 + 1] = rsqrtf(var + 1e-5f);
    }
    __syncthreads();
    for (int e = tid; e < 16 * 256; e += 128) {
      int m = e >> 8, c = e & 255;
      float v = (xf[e] - mv[m * 2]) * mv[m * 2 + 1] * lng[c] + lnb[c];
      xbf[m * 776 + c] = (__bf16)v;
    }
  }
  __syncthreads();

  // proj layer 1, 2 (relu)
  layer16(tid, xbf, 776, 256, p1w, 256, pb1, 1, xf);
  __syncthreads();
  cvt256(tid, xf, xbf);
  __syncthreads();
  layer16(tid, xbf, 776, 256, p2w, 256, pb2, 1, xf);
  __syncthreads();
  cvt256(tid, xf, xbf);
  __syncthreads();

  // proj layer 3 -> global fp32 out [100][8][256]
  for (int t = wave; t < 16; t += 4) {
    const int c0 = t << 4;
    v8f acc = vzero;
    const __bf16* wrow = p3w + (size_t)(c0 + (lane & 15)) * 256;
    for (int kc = 0; kc < 256; kc += 32) {
      v16bf af = fragA(xbf + kc, 776, lane);
      v16bf bf_ = fragB_kcontig(wrow + kc, lane);
      acc = wmma_bf16(af, bf_, acc);
    }
    const int n = lane & 15, mb = (lane >> 4) * 8;
    const float bs = pb3[c0 + n];
#pragma unroll
    for (int r = 0; r < 8; ++r) {
      const int rg = r0 + mb + r;               // rg = q*8 + b, exactly 800 rows
      out[(size_t)rg * 256 + c0 + n] = acc[r] + bs;
    }
  }
}

// ---------------------------------------------------------------------------
// Host launcher
// ---------------------------------------------------------------------------
extern "C" void kernel_launch(void* const* d_in, const int* in_sizes, int n_in,
                              void* d_out, int out_size, void* d_ws, size_t ws_size,
                              hipStream_t stream) {
  (void)in_sizes; (void)n_in; (void)out_size; (void)ws_size;
  const float* output = (const float*)d_in[0];
  const float* feat[3] = {(const float*)d_in[1], (const float*)d_in[2],
                          (const float*)d_in[3]};
  const float* w0 = (const float*)d_in[5];
  const float* w1 = (const float*)d_in[7];
  const float* w2 = (const float*)d_in[9];
  const float* cb[3] = {(const float*)d_in[6], (const float*)d_in[8],
                        (const float*)d_in[10]};
  const float* ln_g = (const float*)d_in[11];
  const float* ln_b = (const float*)d_in[12];
  const float* agg_w1 = (const float*)d_in[13];
  const float* agg_b1 = (const float*)d_in[14];
  const float* agg_w2 = (const float*)d_in[15];
  const float* agg_b2 = (const float*)d_in[16];
  const float* pw[3] = {(const float*)d_in[17], (const float*)d_in[19],
                        (const float*)d_in[21]};
  const float* pb[3] = {(const float*)d_in[18], (const float*)d_in[20],
                        (const float*)d_in[22]};

  const int CIN[3] = {256, 512, 1024};
  const int DD[3]  = {16384, 4096, 1024};
  const int COFF[3] = {0, 256, 768};

  char* ws = (char*)d_ws;
  size_t off = 0;
  auto take = [&](size_t nbytes) -> size_t {
    size_t o = off;
    off = (off + nbytes + 255) & ~(size_t)255;
    return o;
  };
  size_t featbf_o[3], S_o[3];
  for (int l = 0; l < 3; ++l)
    featbf_o[l] = take((size_t)BNUM * CIN[l] * DD[l] * 2);
  size_t obf_o  = take((size_t)BNUM * QP * HID * 2);
  size_t wcat_o = take((size_t)HID * CTOT * 2);
  size_t agg1_o = take((size_t)256 * 768 * 2);
  size_t agg2_o = take((size_t)256 * 256 * 2);
  size_t p1_o   = take((size_t)256 * 256 * 2);
  size_t p2_o   = take((size_t)256 * 256 * 2);
  size_t p3_o   = take((size_t)256 * 256 * 2);
  size_t M_o    = take((size_t)BNUM * QP * CTOT * 2);
  for (int l = 0; l < 3; ++l)
    S_o[l] = take((size_t)BNUM * QP * DD[l] * 2);
  size_t av_o = take((size_t)BNUM * QP * CTOT * 4);

  __bf16* featbf[3];
  for (int l = 0; l < 3; ++l) featbf[l] = (__bf16*)(ws + featbf_o[l]);
  __bf16* obf  = (__bf16*)(ws + obf_o);
  __bf16* wcat = (__bf16*)(ws + wcat_o);
  __bf16* agg1w = (__bf16*)(ws + agg1_o);
  __bf16* agg2w = (__bf16*)(ws + agg2_o);
  __bf16* p1w = (__bf16*)(ws + p1_o);
  __bf16* p2w = (__bf16*)(ws + p2_o);
  __bf16* p3w = (__bf16*)(ws + p3_o);
  __bf16* Mws = (__bf16*)(ws + M_o);
  __bf16* Sws[3];
  for (int l = 0; l < 3; ++l) Sws[l] = (__bf16*)(ws + S_o[l]);
  float* avws = (float*)(ws + av_o);

  // ---- prep: conversions -------------------------------------------------
  for (int l = 0; l < 3; ++l) {
    int n4 = BNUM * CIN[l] * DD[l] / 4;
    cvt_f32_to_bf16<<<dim3((n4 + 255) / 256), 256, 0, stream>>>(feat[l], featbf[l], n4);
  }
  {
    int n = BNUM * QP * HID;
    pad_output_bf16<<<dim3((n + 255) / 256), 256, 0, stream>>>(output, obf);
    int nw = HID * CTOT;
    build_wcat<<<dim3((nw + 255) / 256), 256, 0, stream>>>(w0, w1, w2, wcat);
    int a1 = 256 * 768 / 4;
    cvt_f32_to_bf16<<<dim3((a1 + 255) / 256), 256, 0, stream>>>(agg_w1, agg1w, a1);
    int a2 = 256 * 256 / 4;
    cvt_f32_to_bf16<<<dim3((a2 + 255) / 256), 256, 0, stream>>>(agg_w2, agg2w, a2);
    cvt_f32_to_bf16<<<dim3((a2 + 255) / 256), 256, 0, stream>>>(pw[0], p1w, a2);
    cvt_f32_to_bf16<<<dim3((a2 + 255) / 256), 256, 0, stream>>>(pw[1], p2w, a2);
    cvt_f32_to_bf16<<<dim3((a2 + 255) / 256), 256, 0, stream>>>(pw[2], p3w, a2);
    int nz = BNUM * QP * CTOT;
    zero_f32<<<dim3((nz + 255) / 256), 256, 0, stream>>>(avws, nz);
  }

  // ---- M = O_b @ W_cat : [112 x 1792] per batch --------------------------
  gemm_qp<0, 0><<<dim3(CTOT / 128, BNUM, 1), 128, 0, stream>>>(
      obf, (long long)QP * HID, HID, wcat, 0LL, CTOT, HID,
      (void*)Mws, (long long)QP * CTOT, CTOT);

  // ---- scores S_l = M_l @ feat_l : [112 x D] per (l, b) ------------------
  for (int l = 0; l < 3; ++l) {
    gemm_qp<0, 0><<<dim3(DD[l] / 128, BNUM, 1), 128, 0, stream>>>(
        Mws + COFF[l], (long long)QP * CTOT, CTOT,
        featbf[l], (long long)CIN[l] * DD[l], DD[l], CIN[l],
        (void*)Sws[l], (long long)QP * DD[l], DD[l]);
  }

  // ---- softmax (in place, rows of D) -------------------------------------
  for (int l = 0; l < 3; ++l)
    softmax_rows<<<dim3(QP, BNUM), 256, 0, stream>>>(Sws[l], DD[l]);

  // ---- av = P @ feat^T : split-K (grid.z=16), f32 atomic accumulate ------
  for (int l = 0; l < 3; ++l) {
    gemm_qp<1, 1><<<dim3(CIN[l] / 128, BNUM, 16), 128, 0, stream>>>(
        Sws[l], (long long)QP * DD[l], DD[l],
        featbf[l], (long long)CIN[l] * DD[l], DD[l], DD[l] / 16,
        (void*)(avws + COFF[l]), (long long)QP * CTOT, CTOT);
  }

  // ---- tail: back-project + concat + agg MLP + LN + proj MLP -------------
  tail_kernel<<<dim3(800 / 16), 128, 0, stream>>>(
      avws, wcat, cb[0], cb[1], cb[2], agg1w, agg_b1, agg2w, agg_b2,
      ln_g, ln_b, p1w, pb[0], p2w, pb[1], p3w, pb[2], (float*)d_out);
}
